// MLPBlock_30227979829950
// MI455X (gfx1250) — matrix operand order repaired and checked by reading, not compiled, not run on hardware
//
#include <hip/hip_runtime.h>
#include <hip/hip_bf16.h>

// ---------------------------------------------------------------- constants
constexpr int TDIM = 4096;
constexpr int HDIM = 2048;
constexpr int IDIM = 2048;
constexpr int EDIM = 16;
constexpr int KTOP = 4;

// ---------------------------------------------------------------- types
typedef __attribute__((ext_vector_type(16))) __bf16          v16bf;
typedef __attribute__((ext_vector_type(8)))  float           v8f;
typedef __attribute__((ext_vector_type(16))) unsigned short  v16u16;
typedef __attribute__((ext_vector_type(8)))  unsigned short  u16x8;

// ---------------------------------------------------------------- helpers
__device__ __forceinline__ unsigned short f2bf(float f) {
    unsigned u = __float_as_uint(f);
    unsigned r = u + 0x7FFFu + ((u >> 16) & 1u);   // round-to-nearest-even
    return (unsigned short)(r >> 16);
}
__device__ __forceinline__ float bf2f(unsigned short s) {
    return __uint_as_float(((unsigned)s) << 16);
}
// XOR swizzle of 16B (8-elem) chunks within a 2048-elem LDS row
__device__ __forceinline__ int swz(int row, int col) {
    return ((((col >> 3) ^ (row & 7)) << 3) | (col & 7));
}

// A fragment (16x32 bf16) from a swizzled LDS tile (row stride 2048 elems).
// Row m = lane&15; half h: K = h*16+0..15 (elems 0..15, contiguous)
__device__ __forceinline__ v16bf load_a_lds(const unsigned short* __restrict__ lds,
                                            int row, int k0, int half) {
    u16x8 lo = *(const u16x8*)(lds + row * 2048 + swz(row, k0 + half * 16));
    u16x8 hi = *(const u16x8*)(lds + row * 2048 + swz(row, k0 + half * 16 + 8));
    v16u16 a;
#pragma unroll
    for (int i = 0; i < 8; ++i) { a[i] = lo[i]; a[i + 8] = hi[i]; }
    return __builtin_bit_cast(v16bf, a);
}

// B fragment (32x16 bf16): col n = lane&15 fixed (row pointer passed in),
// element j <-> K = half*16 + j (contiguous along the reduction dim).
template <bool PRE>
__device__ __forceinline__ v16bf load_b(const float* __restrict__ rf,
                                        const unsigned short* __restrict__ rb,
                                        size_t roff, int k) {
    if constexpr (PRE) {
        u16x8 lo = *(const u16x8*)(rb + roff + k);
        u16x8 hi = *(const u16x8*)(rb + roff + k + 8);
        v16u16 b;
#pragma unroll
        for (int i = 0; i < 8; ++i) { b[i] = lo[i]; b[i + 8] = hi[i]; }
        return __builtin_bit_cast(v16bf, b);
    } else {
        const float4* q = (const float4*)(rf + roff + k);
        v16u16 b;
#pragma unroll
        for (int i = 0; i < 4; ++i) {
            float4 v = q[i];
            b[4 * i + 0] = f2bf(v.x);
            b[4 * i + 1] = f2bf(v.y);
            b[4 * i + 2] = f2bf(v.z);
            b[4 * i + 3] = f2bf(v.w);
        }
        return __builtin_bit_cast(v16bf, b);
    }
}

// ---------------------------------------------------------------- RMSNorm
__global__ __launch_bounds__(256) void rmsnorm_kernel(
    const float* __restrict__ x, const float* __restrict__ gamma,
    unsigned short* __restrict__ t_bf) {
    const int t = blockIdx.x, tid = threadIdx.x;
    const int lane = tid & 31, wv = tid >> 5;
    const float* xr = x + (size_t)t * HDIM;
    float v[8];
    float s = 0.f;
#pragma unroll
    for (int j = 0; j < 8; ++j) {
        v[j] = xr[tid + j * 256];
        s += v[j] * v[j];
    }
#pragma unroll
    for (int off = 16; off > 0; off >>= 1) s += __shfl_xor(s, off);
    __shared__ float red[8];
    if (lane == 0) red[wv] = s;
    __syncthreads();
    float tot = 0.f;
#pragma unroll
    for (int i = 0; i < 8; ++i) tot += red[i];
    const float inv = rsqrtf(tot * (1.0f / HDIM) + 1e-6f);
#pragma unroll
    for (int j = 0; j < 8; ++j) {
        int k = tid + j * 256;
        t_bf[(size_t)t * HDIM + k] = f2bf(v[j] * inv * gamma[k]);
    }
}

// ---------------------------------------------------------------- Router
__global__ __launch_bounds__(256) void router_kernel(
    const unsigned short* __restrict__ t_bf, const float* __restrict__ wg,
    const float* __restrict__ bg, int* __restrict__ counts,
    int* __restrict__ tok_idx, float* __restrict__ tok_wt) {
    const int t = blockIdx.x;
    const int tid = threadIdx.x, lane = tid & 31, wv = tid >> 5;
    __shared__ float logits[EDIM];
    const unsigned short* trow = t_bf + (size_t)t * HDIM;
#pragma unroll
    for (int j = 0; j < 2; ++j) {
        const int e = wv * 2 + j;
        const float* wrow = wg + (size_t)e * HDIM;
        float s = 0.f;
        for (int k = lane; k < HDIM; k += 32) s += bf2f(trow[k]) * wrow[k];
#pragma unroll
        for (int off = 16; off > 0; off >>= 1) s += __shfl_xor(s, off);
        if (lane == 0) logits[e] = s + bg[e];
    }
    __syncthreads();
    if (tid == 0) {
        float l[EDIM];
#pragma unroll
        for (int i = 0; i < EDIM; ++i) l[i] = logits[i];
        int sel[KTOP];
        float sv[KTOP];
#pragma unroll
        for (int k = 0; k < KTOP; ++k) {
            int bi = 0;
            float bv = -3.0e38f;
            for (int i = 0; i < EDIM; ++i)
                if (l[i] > bv) { bv = l[i]; bi = i; }
            sel[k] = bi;
            sv[k] = bv;
            l[bi] = -3.0e38f;
        }
        // softmax over top-4 logits == top-k probs renormalized
        const float mx = sv[0];
        float ssum = 0.f, ev[KTOP];
#pragma unroll
        for (int k = 0; k < KTOP; ++k) { ev[k] = __expf(sv[k] - mx); ssum += ev[k]; }
#pragma unroll
        for (int k = 0; k < KTOP; ++k) {
            int pos = atomicAdd(&counts[sel[k]], 1);
            tok_idx[sel[k] * TDIM + pos] = t;
            tok_wt[sel[k] * TDIM + pos] = ev[k] / ssum;
        }
    }
}

// ------------------------------------------------- fp32 -> bf16 weight copy
__global__ __launch_bounds__(256) void cvt_kernel(const float* __restrict__ src,
                                                  unsigned short* __restrict__ dst,
                                                  long long n4) {
    long long i = (long long)blockIdx.x * 256 + threadIdx.x;
    const long long stride = (long long)gridDim.x * 256;
    for (; i < n4; i += stride) {
        float4 v = ((const float4*)src)[i];
        ushort4 o;
        o.x = f2bf(v.x); o.y = f2bf(v.y); o.z = f2bf(v.z); o.w = f2bf(v.w);
        ((ushort4*)dst)[i] = o;
    }
}

// ---------------------------------------------------------------- MoE GEMM
// block = 8 wave32s, handles one (expert, 16-token tile).
template <bool PRE>
__global__ __launch_bounds__(256) void moe_gemm_kernel(
    const float* __restrict__ w1f, const float* __restrict__ w3f,
    const float* __restrict__ w2f, const unsigned short* __restrict__ w1b,
    const unsigned short* __restrict__ w3b, const unsigned short* __restrict__ w2b,
    const unsigned short* __restrict__ t_bf, const int* __restrict__ counts,
    const int* __restrict__ tok_idx, const float* __restrict__ tok_wt,
    float* __restrict__ out) {
    const int e = blockIdx.y;
    const int tile = blockIdx.x;
    const int cnt = counts[e];
    if (tile * 16 >= cnt) return;  // uniform early-exit (EXEC stays all-ones)

    __shared__ unsigned short ldsA[16 * HDIM];  // 64 KB staged t-tile (swizzled)
    __shared__ unsigned short ldsH[16 * IDIM];  // 64 KB h-tile (bf16, swizzled)

    const int tid = threadIdx.x;
    const int lane = tid & 31;
    const int wv = tid >> 5;
    const int cl = lane & 15;
    const int half = lane >> 4;
    const int listbase = e * TDIM + tile * 16;

    // ---- async-stage the gathered 16-token A tile into LDS (swizzled).
    // Each thread copies 16 x 16B chunks of its row; the per-lane LDS dest
    // address lets us apply the XOR bank swizzle during the DMA itself.
    {
        const int row = tid >> 4;    // 0..15
        const int sub = tid & 15;    // 0..15 (16 threads per row)
        const int ii = tile * 16 + row;
        const int tok = (ii < cnt) ? tok_idx[listbase + row] : 0;
        const unsigned short* src = t_bf + (size_t)tok * HDIM;
        const unsigned ldsa_base = (unsigned)(uintptr_t)(&ldsA[0]);
#pragma unroll
        for (int j = 0; j < 16; ++j) {
            const int c = sub + j * 16;                 // 16B chunk index 0..255
            const int csw = c ^ (row & 7);              // swizzled chunk
            const unsigned ldst = ldsa_base + (unsigned)(row * HDIM * 2 + csw * 16);
            const unsigned short* gsrc = src + c * 8;
            asm volatile("global_load_async_to_lds_b128 %0, %1, off"
                         :: "v"(ldst), "v"(gsrc)
                         : "memory");
        }
        asm volatile("s_wait_asynccnt 0" ::: "memory");
    }
    __syncthreads();

    const size_t eb1 = (size_t)e * IDIM * HDIM;
    const float* w1e = w1f + eb1;
    const float* w3e = w3f + eb1;
    const unsigned short* w1eb = PRE ? (w1b + eb1) : nullptr;
    const unsigned short* w3eb = PRE ? (w3b + eb1) : nullptr;

    // ---------------- phase 1: h = silu(t @ w1e^T) * (t @ w3e^T), bf16 -> LDS
    for (int nt = wv; nt < IDIM / 16; nt += 8) {
        const int n = nt * 16 + cl;          // this lane's B column (I index)
        const size_t roff = (size_t)n * HDIM;
        v8f c1 = {};
        v8f c2 = {};
        for (int k0 = 0; k0 < HDIM; k0 += 32) {
            v16bf a = load_a_lds(ldsA, cl, k0, half);
            v16bf b1 = load_b<PRE>(w1e, w1eb, roff, k0 + half * 16);
            v16bf b2 = load_b<PRE>(w3e, w3eb, roff, k0 + half * 16);
            c1 = __builtin_amdgcn_wmma_f32_16x16x32_bf16(false, a, false, b1,
                                                         (short)0, c1, false, false);
            c2 = __builtin_amdgcn_wmma_f32_16x16x32_bf16(false, a, false, b2,
                                                         (short)0, c2, false, false);
        }
#pragma unroll
        for (int r = 0; r < 8; ++r) {
            const float g = c1[r], u = c2[r];
            const float hv = (g / (1.0f + __expf(-g))) * u;   // silu(g)*u
            const int row = half * 8 + r;                     // C layout: M
            ldsH[row * IDIM + swz(row, nt * 16 + cl)] = f2bf(hv);
        }
    }
    __syncthreads();

    // per-lane routing weights / tokens for C rows half*8+r
    int tokr[8];
    float wtr[8];
#pragma unroll
    for (int r = 0; r < 8; ++r) {
        const int row = half * 8 + r;
        const int i = tile * 16 + row;
        if (i < cnt) {
            tokr[r] = tok_idx[listbase + row];
            wtr[r] = tok_wt[listbase + row];
        } else {
            tokr[r] = 0;
            wtr[r] = 0.0f;   // pad rows contribute exactly zero
        }
    }

    const size_t eb2 = (size_t)e * HDIM * IDIM;
    const float* w2e = w2f + eb2;
    const unsigned short* w2eb = PRE ? (w2b + eb2) : nullptr;

    // ---------------- phase 2: out += (h @ w2e^T) * route_weight
    for (int nt = wv; nt < HDIM / 16; nt += 8) {
        const int n = nt * 16 + cl;          // output column (H index)
        const size_t roff = (size_t)n * IDIM;
        v8f acc = {};
        for (int k0 = 0; k0 < IDIM; k0 += 32) {
            v16bf a = load_a_lds(ldsH, cl, k0, half);
            v16bf b = load_b<PRE>(w2e, w2eb, roff, k0 + half * 16);
            acc = __builtin_amdgcn_wmma_f32_16x16x32_bf16(false, a, false, b,
                                                          (short)0, acc, false, false);
        }
#pragma unroll
        for (int r = 0; r < 8; ++r) {
            const float v = acc[r] * wtr[r];
            unsafeAtomicAdd(out + (size_t)tokr[r] * HDIM + n, v);  // global_atomic_add_f32
        }
    }
}

// ---------------------------------------------------------------- launcher
extern "C" void kernel_launch(void* const* d_in, const int* in_sizes, int n_in,
                              void* d_out, int out_size, void* d_ws, size_t ws_size,
                              hipStream_t stream) {
    const float* x     = (const float*)d_in[0];
    const float* gamma = (const float*)d_in[1];
    const float* wg    = (const float*)d_in[2];
    const float* bg    = (const float*)d_in[3];
    const float* w1    = (const float*)d_in[4];
    const float* w3    = (const float*)d_in[5];
    const float* w2    = (const float*)d_in[6];
    float* out = (float*)d_out;

    char* ws = (char*)d_ws;
    size_t off = 0;
    unsigned short* t_bf = (unsigned short*)(ws + off); off += (size_t)TDIM * HDIM * 2;
    int* counts  = (int*)(ws + off);   off += 256;
    int* tok_idx = (int*)(ws + off);   off += (size_t)EDIM * TDIM * 4;
    float* tok_wt = (float*)(ws + off); off += (size_t)EDIM * TDIM * 4;

    const size_t wbytes = (size_t)EDIM * IDIM * HDIM * 2;  // one bf16 weight tensor
    const bool pre = (ws_size >= off + 3 * wbytes);
    unsigned short *w1b = nullptr, *w3b = nullptr, *w2b = nullptr;
    if (pre) {
        w1b = (unsigned short*)(ws + off);
        w3b = (unsigned short*)(ws + off + wbytes);
        w2b = (unsigned short*)(ws + off + 2 * wbytes);
    }

    hipMemsetAsync(counts, 0, 256, stream);
    hipMemcpyAsync(out, x, (size_t)TDIM * HDIM * 4, hipMemcpyDeviceToDevice, stream);

    rmsnorm_kernel<<<TDIM, 256, 0, stream>>>(x, gamma, t_bf);
    router_kernel<<<TDIM, 256, 0, stream>>>(t_bf, wg, bg, counts, tok_idx, tok_wt);

    dim3 grid(TDIM / 16, EDIM);
    if (pre) {
        const long long n4 = (long long)EDIM * IDIM * HDIM / 4;
        cvt_kernel<<<4096, 256, 0, stream>>>(w1, w1b, n4);
        cvt_kernel<<<4096, 256, 0, stream>>>(w3, w3b, n4);
        cvt_kernel<<<4096, 256, 0, stream>>>(w2, w2b, n4);
        moe_gemm_kernel<true><<<grid, 256, 0, stream>>>(w1, w3, w2, w1b, w3b, w2b,
                                                        t_bf, counts, tok_idx, tok_wt, out);
    } else {
        moe_gemm_kernel<false><<<grid, 256, 0, stream>>>(w1, w3, w2, nullptr, nullptr, nullptr,
                                                         t_bf, counts, tok_idx, tok_wt, out);
    }
}
